// KEPCE_GCN_74680891343652
// MI455X (gfx1250) — matrix operand order repaired and checked by reading, not compiled, not run on hardware
//
#include <hip/hip_runtime.h>

typedef __attribute__((ext_vector_type(16))) _Float16 v16h;
typedef __attribute__((ext_vector_type(8)))  _Float16 v8h;
typedef __attribute__((ext_vector_type(8)))  float    v8f;

#define N_NODES_C 100000

// ---------------- degree / normalization ----------------
__global__ void k_deg_init(float* deg, int n) {
  int i = blockIdx.x * blockDim.x + threadIdx.x;
  if (i < n) deg[i] = 1.0f;              // self-loop contributes 1
}

__global__ void k_deg_accum(const int* __restrict__ ei, float* deg, int E) {
  int e = blockIdx.x * blockDim.x + threadIdx.x;
  if (e < E) atomicAdd(&deg[ei[E + e]], 1.0f);
}

__global__ void k_dinv(float* deg, int n) {
  int i = blockIdx.x * blockDim.x + threadIdx.x;
  if (i < n) deg[i] = rsqrtf(deg[i]);    // deg >= 1 always
}

// hconst[c] = sum_i (W0[i]+b0[i]) * W1[i][c]   (layer-0 output is node-independent)
__global__ void k_prep(const float* __restrict__ W0, const float* __restrict__ b0,
                       const float* __restrict__ W1, float* hconst) {
  int c = threadIdx.x;                   // 16 threads
  float s = 0.f;
  for (int i = 0; i < 8; ++i) s += (W0[i] + b0[i]) * W1[i * 16 + c];
  hconst[c] = s;
}

__global__ void k_sacc_init(const float* __restrict__ dinv, float* sacc, int n) {
  int i = blockIdx.x * blockDim.x + threadIdx.x;
  if (i < n) sacc[i] = dinv[i];          // self-loop term of sum(dinv[src])
}

__global__ void k_sacc_accum(const int* __restrict__ ei, const float* __restrict__ dinv,
                             float* sacc, int E) {
  int e = blockIdx.x * blockDim.x + threadIdx.x;
  if (e < E) atomicAdd(&sacc[ei[E + e]], dinv[ei[e]]);
}

// x1 = relu(s1*hconst + b1); g = dinv * (x1 @ W2); acc2 init = g (self loop)
__global__ void k_node1(const float* __restrict__ dinv, const float* __restrict__ sacc,
                        const float* __restrict__ hconst, const float* __restrict__ b1,
                        const float* __restrict__ W2, float* __restrict__ g,
                        float* __restrict__ acc2, int n) {
  int i = blockIdx.x * blockDim.x + threadIdx.x;
  if (i >= n) return;
  float di = dinv[i];
  float s1 = di * sacc[i];
  float x1[16];
#pragma unroll
  for (int c = 0; c < 16; ++c) x1[c] = fmaxf(s1 * hconst[c] + b1[c], 0.f);
#pragma unroll 4
  for (int j = 0; j < 32; ++j) {
    float h = 0.f;
#pragma unroll
    for (int c = 0; c < 16; ++c) h += x1[c] * W2[c * 32 + j];
    float gv = di * h;
    g[i * 32 + j] = gv;
    acc2[i * 32 + j] = gv;
  }
}

// 8 lanes per edge, float4 per lane: acc2[dst] += g[src]
__global__ void k_scatter(const int* __restrict__ ei, const float* __restrict__ g,
                          float* __restrict__ acc2, int E) {
  long tid = (long)blockIdx.x * blockDim.x + threadIdx.x;
  long e = tid >> 3;
  int part = (int)(tid & 7);
  if (e < E) {
    int s = ei[e], d = ei[E + e];
    const float4 v = *(const float4*)(g + (long)s * 32 + part * 4);
    float* base = acc2 + (long)d * 32 + part * 4;
    atomicAdd(base + 0, v.x);
    atomicAdd(base + 1, v.y);
    atomicAdd(base + 2, v.z);
    atomicAdd(base + 3, v.w);
  }
}

// x2 = relu(dinv*acc2 + b2); x3 = relu(x2 @ Wn + bn) -> f16
__global__ void k_node2(const float* __restrict__ dinv, const float* __restrict__ acc2,
                        const float* __restrict__ b2, const float* __restrict__ Wn,
                        const float* __restrict__ bn, _Float16* __restrict__ x3h, int n) {
  int i = blockIdx.x * blockDim.x + threadIdx.x;
  if (i >= n) return;
  float di = dinv[i];
  float x2[32];
#pragma unroll
  for (int j = 0; j < 32; ++j) x2[j] = fmaxf(di * acc2[i * 32 + j] + b2[j], 0.f);
#pragma unroll 4
  for (int k = 0; k < 32; ++k) {
    float s = bn[k];
#pragma unroll
    for (int j = 0; j < 32; ++j) s += x2[j] * Wn[j * 32 + k];
    x3h[i * 32 + k] = (_Float16)fmaxf(s, 0.f);
  }
}

// ---------------- edge MLP: WMMA f16, 16 edges per wave tile ----------------
#define EMLP_WAVES 8

__launch_bounds__(256)
__global__ void k_edge_mlp(const int* __restrict__ ei, const float* __restrict__ ew,
                           const float* __restrict__ ce, const _Float16* __restrict__ x3h,
                           const float* __restrict__ We1, const float* __restrict__ be1,
                           const float* __restrict__ We2, const float* __restrict__ be2,
                           float* __restrict__ out, int E) {
  __shared__ __attribute__((aligned(16))) _Float16 sWe1[96 * 32];   // zero-padded K 66->96
  __shared__ float sbe1[32], sWe2[64], sbe2[2];
  __shared__ __attribute__((aligned(16))) _Float16 sfeat[EMLP_WAVES][16 * 96];
  __shared__ __attribute__((aligned(16))) float    sef[EMLP_WAVES][16 * 32];

  int tid = threadIdx.x;
  for (int idx = tid; idx < 96 * 32; idx += 256) {
    int k = idx >> 5, n = idx & 31;
    sWe1[idx] = (k < 66) ? (_Float16)We1[k * 32 + n] : (_Float16)0.f;
  }
  if (tid < 32) sbe1[tid] = be1[tid];
  if (tid < 64) sWe2[tid] = We2[tid];
  if (tid < 2)  sbe2[tid] = be2[tid];
  __syncthreads();

  const int lane  = tid & 31;
  const int wave  = tid >> 5;
  const int m     = lane & 15;   // edge row within tile / B column / D column
  const int khalf = lane >> 4;   // which K half this lane holds

  // Preload 6 B fragments (We1 as 3 K-chunks x 2 N-tiles), resident in VGPRs.
  v16h B[3][2];
  {
    const int kb = khalf * 16;
#pragma unroll
    for (int kc = 0; kc < 3; ++kc)
#pragma unroll
      for (int nt = 0; nt < 2; ++nt) {
        v16h b;
#pragma unroll
        for (int v = 0; v < 8; ++v) {
          b[2 * v]     = sWe1[(kc * 32 + kb + 2 * v)     * 32 + nt * 16 + m];
          b[2 * v + 1] = sWe1[(kc * 32 + kb + 2 * v + 1) * 32 + nt * 16 + m];
        }
        B[kc][nt] = b;
      }
  }
  const float bias0 = sbe1[m];
  const float bias1 = sbe1[16 + m];
  const float bias2 = sbe2[khalf];
  float w2col[32];
#pragma unroll
  for (int k = 0; k < 32; ++k) w2col[k] = sWe2[k * 2 + khalf];

  _Float16* feat = &sfeat[wave][0];
  float*    ef   = &sef[wave][0];

  const long numTiles  = (long)(E >> 4);
  long t               = (long)blockIdx.x * EMLP_WAVES + wave;
  const long waveTotal = (long)gridDim.x * EMLP_WAVES;

  for (; t < numTiles; t += waveTotal) {
    const long e0 = t << 4;
    const long e  = e0 + m;
    const int s = ei[e], d = ei[(long)E + e];
    _Float16* f = feat + m * 96;
    if (khalf == 0) {
      f[0] = (_Float16)ew[e];
      f[1] = (_Float16)ce[e];
      const unsigned* gs = (const unsigned*)(x3h + (long)s * 32);
      unsigned* dp = (unsigned*)(f + 2);
#pragma unroll
      for (int i = 0; i < 16; ++i) dp[i] = gs[i];
    } else {
      const unsigned* gd = (const unsigned*)(x3h + (long)d * 32);
      unsigned* dp = (unsigned*)(f + 34);
#pragma unroll
      for (int i = 0; i < 16; ++i) dp[i] = gd[i];
      unsigned* zp = (unsigned*)(f + 66);
#pragma unroll
      for (int i = 0; i < 15; ++i) zp[i] = 0u;
    }
    asm volatile("s_wait_dscnt 0" ::: "memory");
    __builtin_amdgcn_wave_barrier();

    v8f acc0, acc1;
#pragma unroll
    for (int v = 0; v < 8; ++v) { acc0[v] = bias0; acc1[v] = bias1; }

    const v8h* arow = (const v8h*)(feat + m * 96);   // 12 groups of 8 halfs
#pragma unroll
    for (int kc = 0; kc < 3; ++kc) {
      union { v16h v; v8h h[2]; } a;
      a.h[0] = arow[kc * 4 + 2 * khalf];       // K = kc*32 + khalf*16 .. +7
      a.h[1] = arow[kc * 4 + 2 * khalf + 1];   // K = kc*32 + khalf*16 + 8 .. +15
      acc0 = __builtin_amdgcn_wmma_f32_16x16x32_f16(false, a.v, false, B[kc][0],
                                                    (short)0, acc0, false, false);
      acc1 = __builtin_amdgcn_wmma_f32_16x16x32_f16(false, a.v, false, B[kc][1],
                                                    (short)0, acc1, false, false);
    }

    // D layout -> row-major ef[16][32] in LDS
    {
      const int rowbase = khalf * 8;
#pragma unroll
      for (int v = 0; v < 8; ++v) {
        ef[(rowbase + v) * 32 + m]      = acc0[v];
        ef[(rowbase + v) * 32 + 16 + m] = acc1[v];
      }
    }
    asm volatile("s_wait_dscnt 0" ::: "memory");
    __builtin_amdgcn_wave_barrier();

    // layer 2: lane (m, khalf) -> edge row m, output column khalf
    {
      float sum = bias2;
      const float* er = ef + m * 32;
#pragma unroll
      for (int k = 0; k < 32; ++k) sum += er[k] * w2col[k];
      out[(e0 + m) * 2 + khalf] = sum;
    }
    asm volatile("s_wait_dscnt 0" ::: "memory");
    __builtin_amdgcn_wave_barrier();
  }
}

// scalar tail (E not multiple of 16) — not used for E=3.2M but kept for safety
__global__ void k_edge_tail(const int* __restrict__ ei, const float* __restrict__ ew,
                            const float* __restrict__ ce, const _Float16* __restrict__ x3h,
                            const float* __restrict__ We1, const float* __restrict__ be1,
                            const float* __restrict__ We2, const float* __restrict__ be2,
                            float* __restrict__ out, int E, int start) {
  int e = start + blockIdx.x * blockDim.x + threadIdx.x;
  if (e >= E) return;
  float f[66];
  f[0] = ew[e]; f[1] = ce[e];
  int s = ei[e], d = ei[E + e];
  for (int i = 0; i < 32; ++i) { f[2 + i] = (float)x3h[s * 32 + i]; f[34 + i] = (float)x3h[d * 32 + i]; }
  float h[32];
  for (int j = 0; j < 32; ++j) {
    float a = be1[j];
    for (int k = 0; k < 66; ++k) a += f[k] * We1[k * 32 + j];
    h[j] = a;
  }
  for (int o = 0; o < 2; ++o) {
    float a = be2[o];
    for (int j = 0; j < 32; ++j) a += h[j] * We2[j * 2 + o];
    out[(long)e * 2 + o] = a;
  }
}

extern "C" void kernel_launch(void* const* d_in, const int* in_sizes, int n_in,
                              void* d_out, int out_size, void* d_ws, size_t ws_size,
                              hipStream_t stream) {
  const int*   ei  = (const int*)  d_in[0];
  const float* ew  = (const float*)d_in[1];
  const float* ce  = (const float*)d_in[2];
  const float* W0  = (const float*)d_in[4];
  const float* b0  = (const float*)d_in[5];
  const float* W1  = (const float*)d_in[6];
  const float* b1  = (const float*)d_in[7];
  const float* W2  = (const float*)d_in[8];
  const float* b2  = (const float*)d_in[9];
  const float* Wn  = (const float*)d_in[10];
  const float* bn  = (const float*)d_in[11];
  const float* We1 = (const float*)d_in[12];
  const float* be1 = (const float*)d_in[13];
  const float* We2 = (const float*)d_in[14];
  const float* be2 = (const float*)d_in[15];
  float* out = (float*)d_out;

  const int E = in_sizes[1];
  const int N = N_NODES_C;

  // workspace carve-out (256B aligned)
  char* ws = (char*)d_ws;
  size_t off = 0;
  auto carve = [&](size_t bytes) { char* p = ws + off; off = (off + bytes + 255) & ~(size_t)255; return p; };
  float*    dinv   = (float*)   carve((size_t)N * 4);
  float*    sacc   = (float*)   carve((size_t)N * 4);
  float*    hconst = (float*)   carve(64);
  float*    g      = (float*)   carve((size_t)N * 32 * 4);
  float*    acc2   = (float*)   carve((size_t)N * 32 * 4);
  _Float16* x3h    = (_Float16*)carve((size_t)N * 32 * 2);
  (void)ws_size; (void)n_in; (void)out_size;

  const int TB = 256;
  k_deg_init <<<(N + TB - 1) / TB, TB, 0, stream>>>(dinv, N);
  k_deg_accum<<<(E + TB - 1) / TB, TB, 0, stream>>>(ei, dinv, E);
  k_dinv     <<<(N + TB - 1) / TB, TB, 0, stream>>>(dinv, N);
  k_prep     <<<1, 16, 0, stream>>>(W0, b0, W1, hconst);
  k_sacc_init<<<(N + TB - 1) / TB, TB, 0, stream>>>(dinv, sacc, N);
  k_sacc_accum<<<(E + TB - 1) / TB, TB, 0, stream>>>(ei, dinv, sacc, E);
  k_node1    <<<(N + 127) / 128, 128, 0, stream>>>(dinv, sacc, hconst, b1, W2, g, acc2, N);
  long sthreads = (long)E * 8;
  k_scatter  <<<(int)((sthreads + TB - 1) / TB), TB, 0, stream>>>(ei, g, acc2, E);
  k_node2    <<<(N + 127) / 128, 128, 0, stream>>>(dinv, acc2, b2, Wn, bn, x3h, N);

  int tiles = E >> 4;
  if (tiles > 0) {
    int blocks = (tiles + EMLP_WAVES - 1) / EMLP_WAVES;
    if (blocks > 4096) blocks = 4096;
    k_edge_mlp<<<blocks, 256, 0, stream>>>(ei, ew, ce, x3h, We1, be1, We2, be2, out, E);
  }
  int tail = E & 15;
  if (tail) {
    k_edge_tail<<<1, tail, 0, stream>>>(ei, ew, ce, x3h, We1, be1, We2, be2, out, E, E - tail);
  }
}
